// GCN_1889785611050
// MI455X (gfx1250) — matrix operand order repaired and checked by reading, not compiled, not run on hardware
//
#include <hip/hip_runtime.h>
#include <stdint.h>

typedef __attribute__((ext_vector_type(16))) _Float16 v16h;
typedef __attribute__((ext_vector_type(8)))  float    v8f;
typedef __attribute__((ext_vector_type(4)))  float    v4f;

#define NN    50000
#define NE    800000
#define CH    128
#define OC    6
#define NG    64

// ---------------- WMMA GEMM: H[N x 128] = X[N x 128] @ W (Wt is W^T in f16) ----
// block = 256 threads = 8 waves; wave w computes columns [16w,16w+16) for the
// block's 16 rows. M = 50000 = 3125 * 16 exactly, so no tail handling.
__global__ __launch_bounds__(256) void k_gemm128(const float* __restrict__ X,
                                                 const _Float16* __restrict__ Wt,
                                                 float* __restrict__ H) {
  const int lane   = threadIdx.x & 31;
  const int wave   = threadIdx.x >> 5;   // N-tile index 0..7
  const int mtile  = blockIdx.x;         // 0..3124
  const int hi     = lane >> 4;          // 0 | 1
  const int l15    = lane & 15;

  const int mA = mtile * 16 + l15;       // A row loaded by this lane
  const int n  = wave * 16 + l15;        // B column owned by this lane

  v8f acc = {};
  #pragma unroll
  for (int ks = 0; ks < 4; ++ks) {       // K = 128 in steps of 32
    // ---- A fragment (16x32 f16): lane<16 -> K 0..15, lane>=16 -> K 16..31
    const int kA = ks * 32 + hi * 16;
    const float* xa = X + (size_t)mA * CH + kA;
    v4f x0 = *(const v4f*)(xa);
    v4f x1 = *(const v4f*)(xa + 4);
    v4f x2 = *(const v4f*)(xa + 8);
    v4f x3 = *(const v4f*)(xa + 12);
    v16h a;
    #pragma unroll
    for (int i = 0; i < 4; ++i) {
      a[i]      = (_Float16)x0[i];
      a[4 + i]  = (_Float16)x1[i];
      a[8 + i]  = (_Float16)x2[i];
      a[12 + i] = (_Float16)x3[i];
    }
    // ---- B fragment (32x16 f16): lane<16 -> K 0..15, lane>=16 -> K 16..31
    const int kB = ks * 32 + hi * 16;
    v16h b = *(const v16h*)(Wt + (size_t)n * CH + kB);

    acc = __builtin_amdgcn_wmma_f32_16x16x32_f16(false, a, false, b,
                                                 (short)0, acc, false, false);
  }
  // ---- store D: lane<16 -> rows m0..m0+7 at col n; lane>=16 -> rows +8
  const int m0 = mtile * 16 + hi * 8;
  float* o = H + (size_t)m0 * CH + n;
  #pragma unroll
  for (int r = 0; r < 8; ++r) o[(size_t)r * CH] = acc[r];
}

// ---------------- weight transpose + f32 -> f16 --------------------------------
__global__ void k_prep_wt(const float* __restrict__ W, _Float16* __restrict__ Wt) {
  int i = blockIdx.x * blockDim.x + threadIdx.x;
  if (i < CH * CH) {
    int k = i / CH, n = i % CH;
    Wt[n * CH + k] = (_Float16)W[k * CH + n];
  }
}

// ---------------- degree / norm -------------------------------------------------
__global__ void k_deg(const int* __restrict__ dst, float* __restrict__ deg) {
  int e = blockIdx.x * blockDim.x + threadIdx.x;
  if (e < NE) atomicAdd(&deg[dst[e]], 1.0f);
}
__global__ void k_dinv(float* __restrict__ d) {
  int i = blockIdx.x * blockDim.x + threadIdx.x;
  if (i < NN) d[i] = rsqrtf(d[i] + 1.0f);
}

// ---------------- edge scatter: agg[dst] += h[src] * dinv[src]*dinv[dst] --------
// one wave per edge, 4 channels per lane (32*4 = 128)
__global__ __launch_bounds__(256) void k_scatter(const int* __restrict__ src,
                                                 const int* __restrict__ dst,
                                                 const float* __restrict__ dinv,
                                                 const float* __restrict__ H,
                                                 float* __restrict__ agg) {
  int gid  = blockIdx.x * blockDim.x + threadIdx.x;
  int e    = gid >> 5;
  int lane = gid & 31;
  if (e >= NE) return;
  int s = src[e], d = dst[e];
  float nsd = dinv[s] * dinv[d];
  const float* hs = H   + (size_t)s * CH + lane * 4;
  float*       ad = agg + (size_t)d * CH + lane * 4;
  v4f v = *(const v4f*)hs;
  atomicAdd(&ad[0], v[0] * nsd);
  atomicAdd(&ad[1], v[1] * nsd);
  atomicAdd(&ad[2], v[2] * nsd);
  atomicAdd(&ad[3], v[3] * nsd);
}

// ---------------- epilogue: act = tanh(agg + h * dinv^2 + bias) -----------------
__global__ void k_epilogue(const float* __restrict__ agg, const float* __restrict__ H,
                           const float* __restrict__ dinv, const float* __restrict__ bias,
                           float* __restrict__ actOut) {
  int i = blockIdx.x * blockDim.x + threadIdx.x;
  if (i >= NN * CH) return;
  int node = i >> 7, c = i & (CH - 1);
  float ns = dinv[node]; ns *= ns;
  actOut[i] = tanhf(agg[i] + H[i] * ns + bias[c]);
}

// ---------------- final 128->6 linear + tanh + pooled sums ----------------------
__global__ void k_final(const float* __restrict__ act, const float* __restrict__ Wl,
                        const float* __restrict__ bl, const int* __restrict__ batch,
                        float* __restrict__ sums, float* __restrict__ cnt) {
  int i = blockIdx.x * blockDim.x + threadIdx.x;
  if (i >= NN) return;
  const float* row = act + (size_t)i * CH;
  float o[OC];
  #pragma unroll
  for (int c = 0; c < OC; ++c) o[c] = bl[c];
  for (int k = 0; k < CH; ++k) {
    float xv = row[k];
    #pragma unroll
    for (int c = 0; c < OC; ++c) o[c] += xv * Wl[k * OC + c];
  }
  int g = batch[i];
  #pragma unroll
  for (int c = 0; c < OC; ++c) atomicAdd(&sums[g * OC + c], tanhf(o[c]));
  atomicAdd(&cnt[g], 1.0f);
}

__global__ void k_pool(const float* __restrict__ sums, const float* __restrict__ cnt,
                       float* __restrict__ out) {
  int i = blockIdx.x * blockDim.x + threadIdx.x;
  if (i >= NG * OC) return;
  out[i] = sums[i] / fmaxf(cnt[i / OC], 1.0f);
}

// -------------------------------------------------------------------------------
extern "C" void kernel_launch(void* const* d_in, const int* in_sizes, int n_in,
                              void* d_out, int out_size, void* d_ws, size_t ws_size,
                              hipStream_t stream) {
  (void)in_sizes; (void)n_in; (void)out_size; (void)ws_size;

  const float* x     = (const float*)d_in[0];
  const int*   ei    = (const int*)d_in[1];
  const int*   src   = ei;
  const int*   dst   = ei + NE;
  const int*   batch = (const int*)d_in[2];
  const float* W[3]  = {(const float*)d_in[3], (const float*)d_in[5], (const float*)d_in[7]};
  const float* B[3]  = {(const float*)d_in[4], (const float*)d_in[6], (const float*)d_in[8]};
  const float* Wl    = (const float*)d_in[9];
  const float* bl    = (const float*)d_in[10];
  float* out = (float*)d_out;

  // workspace carving (256B aligned)
  const size_t szNC = (size_t)NN * CH * sizeof(float);
  uintptr_t p = (uintptr_t)d_ws;
  auto al = [](uintptr_t v) { return (v + 255) & ~(uintptr_t)255; };
  float*    act  = (float*)p;    p = al(p + szNC);
  float*    hbuf = (float*)p;    p = al(p + szNC);
  float*    agg  = (float*)p;    p = al(p + szNC);
  float*    dinv = (float*)p;    p = al(p + (size_t)NN * sizeof(float));
  _Float16* Wt0  = (_Float16*)p; p = al(p + (size_t)CH * CH * sizeof(_Float16));
  _Float16* Wt1  = (_Float16*)p; p = al(p + (size_t)CH * CH * sizeof(_Float16));
  _Float16* Wt2  = (_Float16*)p; p = al(p + (size_t)CH * CH * sizeof(_Float16));
  float*    sums = (float*)p;    // NG*OC floats then NG counts
  float*    cnt  = sums + NG * OC;
  _Float16* Wt[3] = {Wt0, Wt1, Wt2};

  // degree -> dinv
  hipMemsetAsync(dinv, 0, (size_t)NN * sizeof(float), stream);
  k_deg<<<(NE + 255) / 256, 256, 0, stream>>>(dst, dinv);
  k_dinv<<<(NN + 255) / 256, 256, 0, stream>>>(dinv);

  // weight prep (transpose + f16)
  for (int l = 0; l < 3; ++l)
    k_prep_wt<<<(CH * CH + 255) / 256, 256, 0, stream>>>(W[l], Wt[l]);

  // 3 GCN layers
  const float* cur = x;
  for (int l = 0; l < 3; ++l) {
    k_gemm128<<<NN / 16, 256, 0, stream>>>(cur, Wt[l], hbuf);
    hipMemsetAsync(agg, 0, szNC, stream);
    k_scatter<<<(NE * 32 + 255) / 256, 256, 0, stream>>>(src, dst, dinv, hbuf, agg);
    k_epilogue<<<(NN * CH + 255) / 256, 256, 0, stream>>>(agg, hbuf, dinv, B[l], act);
    cur = act;
  }

  // readout + mean pool
  hipMemsetAsync(sums, 0, (size_t)(NG * OC + NG) * sizeof(float), stream);
  k_final<<<(NN + 255) / 256, 256, 0, stream>>>(act, Wl, bl, batch, sums, cnt);
  k_pool<<<(NG * OC + 255) / 256, 256, 0, stream>>>(sums, cnt, out);
}
